// STGN_19988777796044
// MI455X (gfx1250) — matrix operand order repaired and checked by reading, not compiled, not run on hardware
//
#include <hip/hip_runtime.h>
#include <hip/hip_bf16.h>

// ---------------- problem constants ----------------
#define BB    4
#define TT    8
#define NGR   512
#define EE    8192
#define CIN   64
#define HH    128
#define HEADS 4
#define HD    32          // head dim
#define GG    (BB*TT)     // 32 graphs
#define GN    (GG*NGR)    // 16384 nodes
#define ETOT  (GG*EE)     // 262144 edges
#define SS    (TT*NGR)    // 4096 sequence per batch
#define MQ    2           // query tiles per wave in attention
#define AW    8           // waves per attention block

typedef __attribute__((ext_vector_type(16))) _Float16 v16h;
typedef __attribute__((ext_vector_type(8)))  _Float16 v8h;
typedef __attribute__((ext_vector_type(8)))  float    v8f;

__device__ __forceinline__ v8f wmma_f16(v16h a, v16h b, v8f c) {
    return __builtin_amdgcn_wmma_f32_16x16x32_f16(false, a, false, b, (short)0, c,
                                                  false, false);
}

__device__ __forceinline__ v16h cat16(v8h lo, v8h hi) {
    return __builtin_shufflevector(lo, hi, 0, 1, 2, 3, 4, 5, 6, 7,
                                   8, 9, 10, 11, 12, 13, 14, 15);
}

// ---------------- degree / normalization ----------------
__global__ void k_deg_init(float* deg) {
    int i = blockIdx.x * blockDim.x + threadIdx.x;
    if (i < GN) deg[i] = 1.0f;              // self-loop contributes 1
}

__global__ void k_deg_accum(const int* ei, float* deg) {
    int e = blockIdx.x * blockDim.x + threadIdx.x;
    if (e >= ETOT) return;
    int g  = e / EE;
    int le = e - g * EE;
    int dst = ei[g * 2 * EE + EE + le] + g * NGR;
    atomicAdd(&deg[dst], 1.0f);
}

__global__ void k_deg_rsqrt(float* deg) {
    int i = blockIdx.x * blockDim.x + threadIdx.x;
    if (i < GN) deg[i] = rsqrtf(deg[i]);    // in-place -> dinv
}

// ---------------- WMMA GEMM: C[M,N] = A[M,K] @ B (+bias) ----------------
// BT=false: B row-major [K,N].  BT=true: B row-major [N,K], used as B^T.
// grid = (M/16, N/64), block = 128 (4 waves); wave w handles col-tile 4*by+w.
template<int K, bool BT, bool BIAS>
__global__ void __launch_bounds__(128)
k_wmma_gemm(const float* __restrict__ A,
            const float* __restrict__ Bm,
            const float* __restrict__ bias,
            float* __restrict__ C, int Ncols) {
    const int wave  = threadIdx.x >> 5;
    const int lane  = threadIdx.x & 31;
    const int laneM = lane & 15;
    const int half  = lane >> 4;
    const int tm    = blockIdx.x;
    const int tn    = blockIdx.y * 4 + wave;
    const int arow  = tm * 16 + laneM;     // A row this lane feeds
    const int col   = tn * 16 + laneM;     // C column this lane holds

    v8f acc = {};
    #pragma unroll
    for (int k0 = 0; k0 < K; k0 += 32) {
        v16h a, b;
        const float* ar = A + (size_t)arow * K + k0;
        #pragma unroll
        for (int j = 0; j < 8; ++j) a[j]     = (_Float16)ar[half * 8 + j];
        #pragma unroll
        for (int j = 0; j < 8; ++j) a[8 + j] = (_Float16)ar[16 + half * 8 + j];
        #pragma unroll
        for (int j = 0; j < 16; ++j) {
            int kk = k0 + (j < 8 ? half * 8 + j : 16 + half * 8 + (j - 8));
            float v = BT ? Bm[(size_t)col * K + kk]
                         : Bm[(size_t)kk * Ncols + col];
            b[j] = (_Float16)v;
        }
        acc = wmma_f16(a, b, acc);
    }
    float bv = BIAS ? bias[col] : 0.0f;
    #pragma unroll
    for (int v = 0; v < 8; ++v) {
        int r = tm * 16 + v + 8 * half;
        C[(size_t)r * Ncols + col] = acc[v] + bv;
    }
}

// ---------------- GCN epilogue / scatter ----------------
__global__ void k_agg_init(const float* __restrict__ xl,
                           const float* __restrict__ dinv,
                           const float* __restrict__ bias,
                           float* __restrict__ agg) {
    int t = blockIdx.x * blockDim.x + threadIdx.x;
    if (t >= GN * HH) return;
    int node = t >> 7;            // /128
    int c    = t & (HH - 1);
    float di = dinv[node];
    agg[t] = xl[t] * di * di + bias[c];
}

__global__ void k_scatter(const int* __restrict__ ei,
                          const float* __restrict__ xl,
                          const float* __restrict__ dinv,
                          float* __restrict__ agg) {
    long long t = (long long)blockIdx.x * blockDim.x + threadIdx.x;
    if (t >= (long long)ETOT * HH) return;
    int e = (int)(t >> 7);
    int c = (int)(t & (HH - 1));
    int g  = e / EE;
    int le = e - g * EE;
    int src = ei[g * 2 * EE + le]      + g * NGR;
    int dst = ei[g * 2 * EE + EE + le] + g * NGR;
    float nrm = dinv[src] * dinv[dst];
    atomicAdd(&agg[(size_t)dst * HH + c], xl[(size_t)src * HH + c] * nrm);
}

// ---------------- flash attention ----------------
// No-max softmax (scores tiny; exp can't overflow; softmax shift-invariant).
// Block = 256 threads (8 waves) covering 16 query tiles; the block stages the
// 32-key K/V panel to LDS in f16 (V transposed) once per iteration; each wave
// processes MQ=2 query tiles from register-resident fragments.
// grid = (SS/(16*16), B*HEADS) = (16, 16).
__global__ void __launch_bounds__(AW * 32)
k_flash_attn(const float* __restrict__ qkv,
             float* __restrict__ obuf) {
    __shared__ __align__(16) _Float16 kLds[32 * 32];     // [key][d]
    __shared__ __align__(16) _Float16 vLds[32 * 32];     // [d][key] (transposed)
    __shared__ __align__(16) _Float16 plds[AW][16 * 32]; // per-wave P bounce
    const int tid   = threadIdx.x;
    const int wave  = tid >> 5;
    const int lane  = tid & 31;
    const int laneM = lane & 15;
    const int half  = lane >> 4;
    const int b     = blockIdx.y >> 2;
    const int h     = blockIdx.y & (HEADS - 1);
    const int qt0   = blockIdx.x * (AW * MQ) + wave * MQ;  // first q-tile of wave
    const float scale = 0.17677669529663687f;              // 1/sqrt(32)

    // Q fragments (A-layout), pre-scaled
    v16h qa[MQ];
    #pragma unroll
    for (int m = 0; m < MQ; ++m) {
        const float* qr = qkv + (size_t)(b * SS + (qt0 + m) * 16 + laneM) * (3 * HH)
                          + h * HD;
        #pragma unroll
        for (int j = 0; j < 8; ++j) qa[m][j]     = (_Float16)(qr[half * 16 + j] * scale);
        #pragma unroll
        for (int j = 0; j < 8; ++j) qa[m][8 + j] = (_Float16)(qr[half * 16 + 8 + j] * scale);
    }

    v8f o0[MQ], o1[MQ], lrow[MQ];
    #pragma unroll
    for (int m = 0; m < MQ; ++m) {
        o0[m] = (v8f){};
        o1[m] = (v8f){};
        lrow[m] = (v8f){};
    }

    _Float16* myp = plds[wave];
    const v8h* prd = (const v8h*)myp;
    const v8h* kp  = (const v8h*)kLds;
    const v8h* vp  = (const v8h*)vLds;

    // staging mapping: 256 threads, 1024 elements each of K and V:
    // thread t -> key = t/8, d-segment = (t%8)*4 (4 elements each)
    const int skey = tid >> 3;
    const int sd   = (tid & 7) * 4;

    for (int kb = 0; kb < SS; kb += 32) {
        __syncthreads();   // prior iteration's LDS reads complete
        {
            const float* kr = qkv + (size_t)(b * SS + kb + skey) * (3 * HH)
                              + HH + h * HD + sd;
            const float* vr = kr + HH;   // V follows K within the packed row
            #pragma unroll
            for (int j = 0; j < 4; ++j) kLds[skey * 32 + sd + j] = (_Float16)kr[j];
            #pragma unroll
            for (int j = 0; j < 4; ++j) vLds[(sd + j) * 32 + skey] = (_Float16)vr[j];
        }
        __syncthreads();   // staging visible to all waves

        // fragments (all contiguous b128 LDS reads)
        // kf: B[d][key] = K[key][d]: key = laneM (+16), d = half*16 + 0..15
        v16h kf0 = cat16(kp[laneM * 4 + half * 2],        kp[laneM * 4 + half * 2 + 1]);
        v16h kf1 = cat16(kp[(laneM + 16) * 4 + half * 2], kp[(laneM + 16) * 4 + half * 2 + 1]);
        // vf: B[key][d] = VT[d][key]: d = laneM (+16), key = half*16 + 0..15
        v16h vf0 = cat16(vp[laneM * 4 + half * 2],        vp[laneM * 4 + half * 2 + 1]);
        v16h vf1 = cat16(vp[(laneM + 16) * 4 + half * 2], vp[(laneM + 16) * 4 + half * 2 + 1]);

        #pragma unroll
        for (int m = 0; m < MQ; ++m) {
            v8f z = {};
            v8f s0 = wmma_f16(qa[m], kf0, z);   // keys kb..kb+15
            v8f s1 = wmma_f16(qa[m], kf1, z);   // keys kb+16..kb+31
            #pragma unroll
            for (int v = 0; v < 8; ++v) {
                float e0 = __expf(s0[v]);
                float e1 = __expf(s1[v]);
                lrow[m][v] += e0 + e1;
                int r = v + 8 * half;                      // D-layout row
                myp[r * 32 + laneM]      = (_Float16)e0;
                myp[r * 32 + 16 + laneM] = (_Float16)e1;
            }
            v16h pa = cat16(prd[laneM * 4 + half * 2], prd[laneM * 4 + half * 2 + 1]);
            o0[m] = wmma_f16(pa, vf0, o0[m]);   // dims 0..15
            o1[m] = wmma_f16(pa, vf1, o1[m]);   // dims 16..31
        }
    }

    // one final 16-lane reduction of row sums, then normalize + store
    #pragma unroll
    for (int m = 0; m < MQ; ++m) {
        #pragma unroll
        for (int v = 0; v < 8; ++v) {
            float ls = lrow[m][v];
            #pragma unroll
            for (int off = 8; off >= 1; off >>= 1)
                ls += __shfl_xor(ls, off, 32);
            int r    = v + 8 * half;
            int node = b * SS + (qt0 + m) * 16 + r;
            float inv = 1.0f / ls;
            obuf[(size_t)node * HH + h * HD + laneM]      = o0[m][v] * inv;
            obuf[(size_t)node * HH + h * HD + 16 + laneM] = o1[m][v] * inv;
        }
    }
}

// ---------------- host-side orchestration ----------------
extern "C" void kernel_launch(void* const* d_in, const int* in_sizes, int n_in,
                              void* d_out, int out_size, void* d_ws, size_t ws_size,
                              hipStream_t stream) {
    (void)in_sizes; (void)n_in; (void)out_size; (void)ws_size;

    const float* x   = (const float*)d_in[0];
    const int*   ei  = (const int*)  d_in[1];
    const float* W1  = (const float*)d_in[2];
    const float* b1  = (const float*)d_in[3];
    const float* W2  = (const float*)d_in[4];
    const float* b2  = (const float*)d_in[5];
    const float* ipw = (const float*)d_in[6];   // [384,128]
    const float* ipb = (const float*)d_in[7];
    const float* opw = (const float*)d_in[8];   // [128,128]
    const float* opb = (const float*)d_in[9];
    float* out = (float*)d_out;

    float* wsf  = (float*)d_ws;
    float* dinv = wsf;                          // GN
    float* xl   = dinv + GN;                    // GN*HH   (also reused as obuf)
    float* agg  = xl   + (size_t)GN * HH;       // GN*HH   (becomes h1/h2)
    float* qkvb = agg  + (size_t)GN * HH;       // GN*3*HH

    const int T256 = 256;

    // degrees -> dinv
    k_deg_init <<<(GN  + T256 - 1) / T256, T256, 0, stream>>>(dinv);
    k_deg_accum<<<(ETOT + T256 - 1) / T256, T256, 0, stream>>>(ei, dinv);
    k_deg_rsqrt<<<(GN  + T256 - 1) / T256, T256, 0, stream>>>(dinv);

    // GCN layer 1: xl = x @ W1 ; agg = xl*dinv^2 + b1 ; scatter -> h1 (in agg)
    k_wmma_gemm<CIN, false, false><<<dim3(GN / 16, HH / 64), 128, 0, stream>>>(
        x, W1, nullptr, xl, HH);
    k_agg_init<<<(GN * HH) / T256, T256, 0, stream>>>(xl, dinv, b1, agg);
    k_scatter <<<((long long)ETOT * HH) / T256, T256, 0, stream>>>(ei, xl, dinv, agg);

    // GCN layer 2: xl = h1 @ W2 ; agg = xl*dinv^2 + b2 ; scatter -> h2 (in agg)
    k_wmma_gemm<HH, false, false><<<dim3(GN / 16, HH / 64), 128, 0, stream>>>(
        agg, W2, nullptr, xl, HH);
    k_agg_init<<<(GN * HH) / T256, T256, 0, stream>>>(xl, dinv, b2, agg);
    k_scatter <<<((long long)ETOT * HH) / T256, T256, 0, stream>>>(ei, xl, dinv, agg);

    // qkv = h2 @ in_proj_w^T + in_proj_b
    k_wmma_gemm<HH, true, true><<<dim3(GN / 16, (3 * HH) / 64), 128, 0, stream>>>(
        agg, ipw, ipb, qkvb, 3 * HH);

    // flash attention -> obuf (reuses xl)
    k_flash_attn<<<dim3(SS / (16 * AW * MQ), BB * HEADS), AW * 32, 0, stream>>>(qkvb, xl);

    // out = obuf @ out_proj_w^T + out_proj_b
    k_wmma_gemm<HH, true, true><<<dim3(GN / 16, HH / 64), 128, 0, stream>>>(
        xl, opw, opb, out, HH);
}
